// VarianceAdaptor_75685913690790
// MI455X (gfx1250) — matrix operand order repaired and checked by reading, not compiled, not run on hardware
//
#include <hip/hip_runtime.h>

// ---------- types ----------
typedef __attribute__((ext_vector_type(16))) __bf16 v16bf;
typedef __attribute__((ext_vector_type(8)))  float  v8f;
struct B32x { uint4 lo, hi; };   // 32 bytes == one v16bf fragment

__device__ __forceinline__ unsigned short f2bf(float f) {
    unsigned int u = __float_as_uint(f);
    u += 0x7FFFu + ((u >> 16) & 1u);          // round-to-nearest-even
    return (unsigned short)(u >> 16);
}
__device__ __forceinline__ float bf2f(unsigned short h) {
    return __uint_as_float(((unsigned int)h) << 16);
}

// ---------- constants ----------
#define BATCH 16
#define TSRC  512
#define TMEL  4096
#define CCH   256     // channels (Cin == Cout == 256)
#define KTOT  768     // 3 * 256 conv-as-gemm reduction
#define KSN   24      // KTOT / 32 wmma K-steps

// =====================================================================
// Pack conv weight (3,256,256) fp32 -> bf16 in WMMA-B layout:
//   wp[((ks*16 + nt)*16 + n)*32 + kk],  Kg = ks*32+kk, N = nt*16+n
// =====================================================================
__global__ void pack_w_kernel(const float* __restrict__ w,
                              unsigned short* __restrict__ wp) {
    int id = blockIdx.x * 256 + threadIdx.x;          // over KTOT*CCH
    if (id >= KTOT * CCH) return;
    int Kg = id >> 8, N = id & 255;
    int ks = Kg >> 5, kk = Kg & 31;
    int nt = N >> 4,  n  = N & 15;
    // (dk,cin,cout) flat == Kg*256 + N
    wp[((ks * 16 + nt) * 16 + n) * 32 + kk] = f2bf(w[Kg * 256 + N]);
}

// =====================================================================
// fp32 -> bf16 bulk convert
// =====================================================================
__global__ void f2bf_kernel(const float* __restrict__ in,
                            unsigned short* __restrict__ out, int n) {
    int i = blockIdx.x * 256 + threadIdx.x;
    if (i < n) out[i] = f2bf(in[i]);
}

// =====================================================================
// Per-batch cumsum of duration + mel_len (tiny)
// =====================================================================
__global__ void cumsum_kernel(const int* __restrict__ dur,
                              int* __restrict__ cum,
                              float* __restrict__ mel_out,
                              const int* __restrict__ maxlen_p) {
    int b = threadIdx.x;
    if (b >= BATCH) return;
    int ml = maxlen_p[0];
    const int* d = dur + b * TSRC;
    int* c = cum + b * TSRC;
    int s = 0;
    for (int t = 0; t < TSRC; ++t) { s += d[t]; c[t] = s; }
    mel_out[b] = (float)(s < ml ? s : ml);
}

// =====================================================================
// Length regulator gather: xe (fp32 into d_out) + bf16 copy for predictors
// =====================================================================
__global__ void gather_kernel(const float* __restrict__ x,
                              const int* __restrict__ cum,
                              float* __restrict__ xe,
                              unsigned short* __restrict__ xebf,
                              const int* __restrict__ maxlen_p) {
    int b = blockIdx.y, t = blockIdx.x, c = threadIdx.x;
    const int* cb = cum + b * TSRC;
    // searchsorted(cum, t, side='right'): first i with cum[i] > t
    int lo = 0, hi = TSRC;
    while (lo < hi) { int mid = (lo + hi) >> 1; if (cb[mid] > t) hi = mid; else lo = mid + 1; }
    int idx = lo < (TSRC - 1) ? lo : (TSRC - 1);
    int ml = maxlen_p[0];
    int mell = cb[TSRC - 1] < ml ? cb[TSRC - 1] : ml;
    float v = (t < mell) ? x[((size_t)b * TSRC + idx) * CCH + c] : 0.f;
    size_t o = ((size_t)b * TMEL + t) * CCH + c;
    xe[o] = v;
    xebf[o] = f2bf(v);
}

// =====================================================================
// Fused conv1d(K=3, 256->256, SAME) + bias + ReLU + LayerNorm, bf16 out.
// Block = 256 threads (8 waves), one (batch, 32-row time tile).
// Wave w owns N-tiles {2w, 2w+1} x M-tiles {0,1}: 4 accumulators;
// each A and B fragment is reused by 2 wmma (2 loads per wmma).
// Halo staging uses async global->LDS copies (ASYNCcnt path).
// =====================================================================
__global__ __launch_bounds__(256)
void conv_ln_kernel(const unsigned short* __restrict__ in,
                    const unsigned short* __restrict__ wpack,
                    const float* __restrict__ bias,
                    const float* __restrict__ gamma,
                    const float* __restrict__ beta,
                    unsigned short* __restrict__ outbf, int T) {
    const int XSTR = 264;   // bf16 row stride (pad: bank-conflict-free b128)
    const int HSTR = 260;   // fp32 row stride
    __shared__ __align__(16) unsigned short xin[34 * XSTR];
    __shared__ __align__(16) float hbuf[32 * HSTR];
    __shared__ float reds[32][17], redq[32][17];
    __shared__ float s_mean[32], s_rstd[32];

    const int b  = blockIdx.y;
    const int t0 = blockIdx.x * 32;
    const int tid = threadIdx.x;
    const unsigned short* inb = in + (size_t)b * T * CCH;

    // ---- stage 34 halo rows (t0-1 .. t0+32) as bf16, zero-padded ----
    // 16 bytes (8 bf16) per chunk; async copy global -> LDS for in-range
    // rows, plain zero store for the padded boundary rows.
    for (int i = tid; i < 34 * 32; i += 256) {
        int row = i >> 5, c8 = (i & 31) * 8;
        int t = t0 - 1 + row;
        unsigned short* lp = xin + row * XSTR + c8;
        if (t >= 0 && t < T) {
            unsigned lds_off = (unsigned)(uintptr_t)lp;
            const unsigned short* g = inb + (size_t)t * CCH + c8;
            asm volatile("global_load_async_to_lds_b128 %0, %1, off"
                         :: "v"(lds_off), "v"(g) : "memory");
        } else {
            uint4 z = {0u, 0u, 0u, 0u};
            *(uint4*)lp = z;
        }
    }
    asm volatile("s_wait_asynccnt 0x0" ::: "memory");
    __syncthreads();

    const int wave = tid >> 5, lane = tid & 31;
    const int nlo   = lane & 15;
    const int khalf = (lane < 16) ? 0 : 8;    // A-matrix K base per half-wave
    const int kbase = (lane < 16) ? 0 : 8;    // B-matrix K base per half-wave
    const int moff  = (lane < 16) ? 0 : 8;    // C/D row offset per half-wave
    const int nt0 = wave * 2, nt1 = wave * 2 + 1;

    v8f acc00 = {0.f,0.f,0.f,0.f,0.f,0.f,0.f,0.f};
    v8f acc01 = acc00, acc10 = acc00, acc11 = acc00;

#pragma unroll 2
    for (int ks = 0; ks < KSN; ++ks) {
        int dk = ks >> 3;              // which of the 3 conv taps
        int c0 = (ks & 7) * 32;        // channel offset inside the tap
        // A fragments (M-tile 0: rows 0-15, M-tile 1: rows 16-31)
        const unsigned short* a0 = xin + (nlo + dk) * XSTR + c0 + khalf;
        const unsigned short* a1 = a0 + 16 * XSTR;
        B32x au0, au1;
        au0.lo = *(const uint4*)(a0); au0.hi = *(const uint4*)(a0 + 16);
        au1.lo = *(const uint4*)(a1); au1.hi = *(const uint4*)(a1 + 16);
        v16bf A0 = __builtin_bit_cast(v16bf, au0);
        v16bf A1 = __builtin_bit_cast(v16bf, au1);
        // B fragments: two 16-byte chunks per lane (kk kbase..+7, kbase+16..+23) from packed weights
        const unsigned short* w0 = wpack + (((ks * 16 + nt0) * 16 + nlo) * 32 + kbase);
        const unsigned short* w1 = wpack + (((ks * 16 + nt1) * 16 + nlo) * 32 + kbase);
        B32x bu0, bu1;
        bu0.lo = *(const uint4*)(w0); bu0.hi = *(const uint4*)(w0 + 16);
        bu1.lo = *(const uint4*)(w1); bu1.hi = *(const uint4*)(w1 + 16);
        v16bf B0 = __builtin_bit_cast(v16bf, bu0);
        v16bf B1 = __builtin_bit_cast(v16bf, bu1);
        acc00 = __builtin_amdgcn_wmma_f32_16x16x32_bf16(false, A0, false, B0,
                                                        (short)0, acc00, false, false);
        acc01 = __builtin_amdgcn_wmma_f32_16x16x32_bf16(false, A0, false, B1,
                                                        (short)0, acc01, false, false);
        acc10 = __builtin_amdgcn_wmma_f32_16x16x32_bf16(false, A1, false, B0,
                                                        (short)0, acc10, false, false);
        acc11 = __builtin_amdgcn_wmma_f32_16x16x32_bf16(false, A1, false, B1,
                                                        (short)0, acc11, false, false);
    }

    // ---- bias + ReLU, spill tiles to LDS fp32 ----
    const int ch0 = nt0 * 16 + nlo, ch1 = nt1 * 16 + nlo;
    const float bv0 = bias[ch0], bv1 = bias[ch1];
#pragma unroll
    for (int r = 0; r < 8; ++r) {
        float v;
        v = acc00[r] + bv0; hbuf[(r + moff) * HSTR + ch0]        = v > 0.f ? v : 0.f;
        v = acc01[r] + bv1; hbuf[(r + moff) * HSTR + ch1]        = v > 0.f ? v : 0.f;
        v = acc10[r] + bv0; hbuf[(16 + r + moff) * HSTR + ch0]   = v > 0.f ? v : 0.f;
        v = acc11[r] + bv1; hbuf[(16 + r + moff) * HSTR + ch1]   = v > 0.f ? v : 0.f;
    }
    __syncthreads();

    // ---- LayerNorm over 256 channels per time row (32 rows) ----
    const int seg = tid & 15;
#pragma unroll
    for (int rr = 0; rr < 2; ++rr) {
        int row = (tid >> 4) + rr * 16;
        float s = 0.f, q = 0.f;
#pragma unroll
        for (int j = 0; j < 16; ++j) {
            float v = hbuf[row * HSTR + seg * 16 + j];
            s += v; q += v * v;
        }
        reds[row][seg] = s; redq[row][seg] = q;
    }
    __syncthreads();
    if (seg == 0) {
#pragma unroll
        for (int rr = 0; rr < 2; ++rr) {
            int row = (tid >> 4) + rr * 16;
            float ts = 0.f, tq = 0.f;
#pragma unroll
            for (int j = 0; j < 16; ++j) { ts += reds[row][j]; tq += redq[row][j]; }
            float mean = ts * (1.f / 256.f);
            float var  = tq * (1.f / 256.f) - mean * mean;
            s_mean[row] = mean;
            s_rstd[row] = rsqrtf(var + 1e-5f);
        }
    }
    __syncthreads();

#pragma unroll
    for (int rr = 0; rr < 2; ++rr) {
        int row = (tid >> 4) + rr * 16;
        const float mean = s_mean[row], rstd = s_rstd[row];
        unsigned short* ob = outbf + ((size_t)b * T + t0 + row) * CCH;
#pragma unroll
        for (int j = 0; j < 16; ++j) {
            int c = seg * 16 + j;
            float v = (hbuf[row * HSTR + c] - mean) * rstd * gamma[c] + beta[c];
            ob[c] = f2bf(v);
        }
    }
}

// =====================================================================
// Linear head: pred[b,t] = sum_c h[b,t,c]*wl[c] + bl   (uint4 = 8 bf16)
// =====================================================================
__global__ void linear_kernel(const unsigned short* __restrict__ h,
                              const float* __restrict__ wl,
                              const float* __restrict__ bl,
                              float* __restrict__ out, int T) {
    int i = blockIdx.x * 256 + threadIdx.x;
    if (i >= BATCH * T) return;
    const uint4* r = (const uint4*)(h + (size_t)i * CCH);
    float s = bl[0];
    for (int cc = 0; cc < CCH / 8; ++cc) {
        uint4 u = r[cc];
        const unsigned int w[4] = {u.x, u.y, u.z, u.w};
#pragma unroll
        for (int j = 0; j < 4; ++j) {
            int c = cc * 8 + j * 2;
            s += bf2f((unsigned short)(w[j] & 0xFFFFu)) * wl[c];
            s += bf2f((unsigned short)(w[j] >> 16))     * wl[c + 1];
        }
    }
    out[i] = s;
}

// =====================================================================
// Pitch/energy embedding convs (scalar input, K=3) + in-place final add
// =====================================================================
__global__ void fuse_out_kernel(const float* __restrict__ pt,
                                const float* __restrict__ et,
                                const float* __restrict__ pw,
                                const float* __restrict__ pb,
                                const float* __restrict__ ew,
                                const float* __restrict__ eb,
                                float* __restrict__ out) {
    int b = blockIdx.y, t = blockIdx.x, c = threadIdx.x;
    const float* pr = pt + (size_t)b * TMEL;
    const float* er = et + (size_t)b * TMEL;
    float pm1 = (t > 0) ? pr[t - 1] : 0.f, p0 = pr[t], pp1 = (t < TMEL - 1) ? pr[t + 1] : 0.f;
    float em1 = (t > 0) ? er[t - 1] : 0.f, e0 = er[t], ep1 = (t < TMEL - 1) ? er[t + 1] : 0.f;
    float pe = pb[c] + pw[c] * pm1 + pw[256 + c] * p0 + pw[512 + c] * pp1;
    float ee = eb[c] + ew[c] * em1 + ew[256 + c] * e0 + ew[512 + c] * ep1;
    size_t o = ((size_t)b * TMEL + t) * CCH + c;
    out[o] += pe + ee;
}

// =====================================================================
// Launch
// =====================================================================
extern "C" void kernel_launch(void* const* d_in, const int* in_sizes, int n_in,
                              void* d_out, int out_size, void* d_ws, size_t ws_size,
                              hipStream_t stream) {
    (void)in_sizes; (void)n_in; (void)out_size; (void)ws_size;

    const float* x        = (const float*)d_in[0];
    const int*   dur_tgt  = (const int*)d_in[2];
    const float* pitch_t  = (const float*)d_in[3];
    const float* energy_t = (const float*)d_in[4];
    const int*   maxlen_p = (const int*)d_in[5];
    // predictor params: base index P -> w1,b1,g1,beta1,w2,b2,g2,beta2,wl,bl
    const int PD = 6, PP = 16, PE = 26;
    const float* pemb_w = (const float*)d_in[36];
    const float* pemb_b = (const float*)d_in[37];
    const float* eemb_w = (const float*)d_in[38];
    const float* eemb_b = (const float*)d_in[39];

    // ---- d_out layout (tuple concatenated flat) ----
    float* out_main  = (float*)d_out;                         // (16,4096,256)
    float* out_mel   = out_main + (size_t)BATCH * TMEL * CCH; // (16,)
    float* out_dur   = out_mel + BATCH;                       // (16,512)
    float* out_pitch = out_dur + BATCH * TSRC;                // (16,4096)
    float* out_energy= out_pitch + BATCH * TMEL;              // (16,4096)

    // ---- workspace carve-up ----
    char* W = (char*)d_ws;
    size_t o = 0;
    auto carve = [&](size_t bytes) {
        void* p = W + o;
        o += bytes; o = (o + 255) & ~(size_t)255;
        return p;
    };
    unsigned short* xbf   = (unsigned short*)carve((size_t)BATCH * TSRC * CCH * 2);
    unsigned short* xebf  = (unsigned short*)carve((size_t)BATCH * TMEL * CCH * 2);
    unsigned short* hA    = (unsigned short*)carve((size_t)BATCH * TMEL * CCH * 2);
    unsigned short* hB    = (unsigned short*)carve((size_t)BATCH * TMEL * CCH * 2);
    int*            cum   = (int*)carve((size_t)BATCH * TSRC * 4);
    const size_t WPB = (size_t)KTOT * CCH * 2;  // 393216 bytes per packed weight
    unsigned short* wp[6];
    for (int i = 0; i < 6; ++i) wp[i] = (unsigned short*)carve(WPB);

    // ---- 1. pack conv weights to WMMA-B bf16 layout ----
    const int wsrc[6] = { PD + 0, PD + 4, PP + 0, PP + 4, PE + 0, PE + 4 };
    for (int i = 0; i < 6; ++i)
        pack_w_kernel<<<(KTOT * CCH + 255) / 256, 256, 0, stream>>>(
            (const float*)d_in[wsrc[i]], wp[i]);

    // ---- 2. x -> bf16 ----
    {
        int n = BATCH * TSRC * CCH;
        f2bf_kernel<<<(n + 255) / 256, 256, 0, stream>>>(x, xbf, n);
    }

    // ---- 3. cumsum + mel_len ----
    cumsum_kernel<<<1, 32, 0, stream>>>(dur_tgt, cum, out_mel, maxlen_p);

    // ---- 4. length regulator: xe (fp32 into d_out) + bf16 ----
    gather_kernel<<<dim3(TMEL, BATCH), 256, 0, stream>>>(x, cum, out_main, xebf, maxlen_p);

    // ---- 5. three variance predictors ----
    auto predictor = [&](const unsigned short* input, int T, int P,
                         unsigned short* w1p, unsigned short* w2p, float* pred) {
        dim3 grid(T / 32, BATCH);
        conv_ln_kernel<<<grid, 256, 0, stream>>>(
            input, w1p, (const float*)d_in[P + 1], (const float*)d_in[P + 2],
            (const float*)d_in[P + 3], hA, T);
        conv_ln_kernel<<<grid, 256, 0, stream>>>(
            hA, w2p, (const float*)d_in[P + 5], (const float*)d_in[P + 6],
            (const float*)d_in[P + 7], hB, T);
        linear_kernel<<<(BATCH * T + 255) / 256, 256, 0, stream>>>(
            hB, (const float*)d_in[P + 8], (const float*)d_in[P + 9], pred, T);
    };
    predictor(xbf,  TSRC, PD, wp[0], wp[1], out_dur);
    predictor(xebf, TMEL, PP, wp[2], wp[3], out_pitch);
    predictor(xebf, TMEL, PE, wp[4], wp[5], out_energy);

    // ---- 6. embedding convs + final add (in place on d_out) ----
    fuse_out_kernel<<<dim3(TMEL, BATCH), 256, 0, stream>>>(
        pitch_t, energy_t, pemb_w, pemb_b, eemb_w, eemb_b, out_main);
}
